// DenseCapsuleLayer_44753559225144
// MI455X (gfx1250) — matrix operand order repaired and checked by reading, not compiled, not run on hardware
//
#include <hip/hip_runtime.h>
#include <hip/hip_bf16.h>

typedef __bf16 bf16;
typedef __attribute__((ext_vector_type(16))) __bf16 v16bf;
typedef __attribute__((ext_vector_type(8)))  __bf16 v8bf;
typedef __attribute__((ext_vector_type(8)))  float  v8f;

#define CB   32      // batch
#define CO   64      // out capsules
#define CI   2048    // in capsules
#define CVO  32      // out vec dim
#define CVI  16      // in vec dim
#define NPART 16
#define IPP  (CI / NPART)   // 128 i's per wave

// ---------------------------------------------------------------------------
// Convert W fp32 [O][I][VO][VI] -> bf16 same layout (s-pass B operand)
// and bf16 transposed [O][I][VI][VO] (agreement-pass A operand).
// ---------------------------------------------------------------------------
__global__ void caps_conv_w(const float* __restrict__ W,
                            bf16* __restrict__ Wbf, bf16* __restrict__ WT) {
    const int total = CO * CI * CVO * CVI;  // 67,108,864
    for (int idx = blockIdx.x * blockDim.x + threadIdx.x; idx < total;
         idx += gridDim.x * blockDim.x) {
        int u  = idx & (CVI - 1);
        int v  = (idx >> 4) & (CVO - 1);
        int oi = idx >> 9;                  // o*CI + i
        bf16 wb = (bf16)W[idx];
        Wbf[idx] = wb;
        WT[((long)oi * CVI + u) * CVO + v] = wb;
    }
}

// x fp32 [B][I][VI] -> bf16 [B][I][VI]
__global__ void caps_conv_x(const float* __restrict__ x, bf16* __restrict__ x1) {
    const int total = CB * CI * CVI;        // 1,048,576
    for (int idx = blockIdx.x * blockDim.x + threadIdx.x; idx < total;
         idx += gridDim.x * blockDim.x) {
        x1[idx] = (bf16)x[idx];
    }
}

// ---------------------------------------------------------------------------
// s-pass:  S_o[b][v] = sum_{i,u} c[b,o,i]*x[b,i,u] * W[o,i,v,u]
// One wave per (o, I-partition). K-chunk of 32 = two i's.
// UNIFORM=1: c == 1/64 commutes out of the GEMM -> scale accumulator at end.
// ---------------------------------------------------------------------------
template <int UNIFORM>
__global__ __launch_bounds__(32)
void caps_spass(const bf16* __restrict__ Wbf, const bf16* __restrict__ x1,
                const bf16* __restrict__ cT, float* __restrict__ s_part) {
    const int o    = blockIdx.x / NPART;
    const int part = blockIdx.x % NPART;
    const int lane = threadIdx.x;
    const int h = lane >> 4, m = lane & 15;

    v8f acc00 = {}, acc01 = {}, acc10 = {}, acc11 = {};
    const bf16* Wo = Wbf + (long)o * CI * CVO * CVI;
    const int i_base = part * IPP;

    for (int ii = 0; ii < IPP; ii += 2) {
        const int i0 = i_base + ii, i1 = i0 + 1;
        const int ib = i0 + h;          // this lane-half carries K of i_h (pure ALU)
        // B fragments: one full W row (16 u's) per lane, per N-half
        v16bf bfrag0 = *(const v16bf*)(Wo + ((long)ib * CVO + m) * CVI);
        v16bf bfrag1 = *(const v16bf*)(Wo + ((long)ib * CVO + m + 16) * CVI);
        // unconditional speculative prefetch ~8 i's ahead (harmless past range)
        __builtin_prefetch(Wo + ((long)(ib + 8) * CVO + m) * CVI, 0, 1);
        __builtin_prefetch(Wo + ((long)(ib + 8) * CVO + m + 16) * CVI, 0, 1);

        #pragma unroll
        for (int mh = 0; mh < 2; ++mh) {
            const int b = m + 16 * mh;
            v8bf xv0 = *(const v8bf*)(x1 + ((long)b * CI + ib) * CVI);
            v8bf xv1 = *(const v8bf*)(x1 + ((long)b * CI + ib) * CVI + 8);
            v16bf afrag;
            if (UNIFORM) {
                #pragma unroll
                for (int e = 0; e < 8; ++e) { afrag[e] = xv0[e]; afrag[e + 8] = xv1[e]; }
            } else {
                float c0 = (float)cT[((long)o * CI + ib) * CB + b];
                float c1 = (float)cT[((long)o * CI + ib) * CB + b];
                #pragma unroll
                for (int e = 0; e < 8; ++e) {
                    afrag[e]     = (bf16)((float)xv0[e] * c0);
                    afrag[e + 8] = (bf16)((float)xv1[e] * c1);
                }
            }
            if (mh == 0) {
                acc00 = __builtin_amdgcn_wmma_f32_16x16x32_bf16(false, afrag, false, bfrag0, (short)0, acc00, false, false);
                acc01 = __builtin_amdgcn_wmma_f32_16x16x32_bf16(false, afrag, false, bfrag1, (short)0, acc01, false, false);
            } else {
                acc10 = __builtin_amdgcn_wmma_f32_16x16x32_bf16(false, afrag, false, bfrag0, (short)0, acc10, false, false);
                acc11 = __builtin_amdgcn_wmma_f32_16x16x32_bf16(false, afrag, false, bfrag1, (short)0, acc11, false, false);
            }
        }
    }
    // D layout: VGPR r -> row = r + 8h (+16*mh), col = m (+16*nh)
    const float cs = UNIFORM ? 0.015625f : 1.0f;
    float* sp = s_part + (long)(o * NPART + part) * CB * CVO;
    #pragma unroll
    for (int r = 0; r < 8; ++r) {
        int row0 = 8 * h + r;
        sp[(row0)      * CVO + m]      = acc00[r] * cs;
        sp[(row0)      * CVO + m + 16] = acc01[r] * cs;
        sp[(row0 + 16) * CVO + m]      = acc10[r] * cs;
        sp[(row0 + 16) * CVO + m + 16] = acc11[r] * cs;
    }
}

// ---------------------------------------------------------------------------
// agreement pass (transposed): T'[u][b] = sum_v W[o,i,v,u] * v[b,o,v]
//   A = WT row (M=u, K=v), B = V (K=v, N=b half) -> lane holds 8 u's of ONE b.
// dot[b,o,i] = sum_u T'[u][b]*x[b,i,u]: 8 FMAs + one shfl_xor(16).
// blog[b][i][o] (+)= dot.  ACCUM templated; single predicated store region.
// ---------------------------------------------------------------------------
template <int ACCUM>
__global__ __launch_bounds__(32)
void caps_apass(const bf16* __restrict__ WT, const bf16* __restrict__ x1,
                const bf16* __restrict__ vbf, float* __restrict__ blog) {
    const int o    = blockIdx.x / NPART;
    const int part = blockIdx.x % NPART;
    const int lane = threadIdx.x;
    const int h = lane >> 4, n = lane & 15;

    // B fragments (V matrix: K=v rows, N=b cols) — loop-invariant, per b-half
    v16bf bfrag[2];
    #pragma unroll
    for (int nh = 0; nh < 2; ++nh) {
        const int b = n + 16 * nh;
        bfrag[nh] = *(const v16bf*)(vbf + ((long)o * CB + b) * CVO + 16 * h);
    }

    for (int ii = 0; ii < IPP; ++ii) {
        const int i = part * IPP + ii;
        // A fragment: lane (h, m=n): row u=n of W_{o,i}^T, v in {16h..16h+15}
        const bf16* wrow = WT + ((long)(o * CI + i) * CVI + n) * CVO;
        v8bf alo = *(const v8bf*)(wrow + 16 * h);
        v8bf ahi = *(const v8bf*)(wrow + 16 * h + 8);
        v16bf afrag;
        #pragma unroll
        for (int e = 0; e < 8; ++e) { afrag[e] = alo[e]; afrag[e + 8] = ahi[e]; }
        __builtin_prefetch(WT + ((long)(o * CI + i + 8) * CVI + n) * CVO, 0, 1);

        // both b-halves first (no branches), then one predicated store region
        float dot[2];
        #pragma unroll
        for (int nh = 0; nh < 2; ++nh) {
            v8f t = {};
            t = __builtin_amdgcn_wmma_f32_16x16x32_bf16(false, afrag, false, bfrag[nh], (short)0, t, false, false);
            const int b = n + 16 * nh;
            v8bf xv = *(const v8bf*)(x1 + ((long)b * CI + i) * CVI + 8 * h);
            float p = 0.0f;
            #pragma unroll
            for (int r = 0; r < 8; ++r) p += t[r] * (float)xv[r];
            dot[nh] = p + __shfl_xor(p, 16, 32);   // merge u-halves
        }
        if (h == 0) {
            long idx0 = ((long)(n)      * CI + i) * CO + o;
            long idx1 = ((long)(n + 16) * CI + i) * CO + o;
            if (ACCUM) {
                blog[idx0] += dot[0];
                blog[idx1] += dot[1];
            } else {
                blog[idx0] = dot[0];
                blog[idx1] = dot[1];
            }
        }
    }
}

// softmax over O for each (b,i); write c transposed [O][I][B] in bf16
__global__ void caps_softmax(const float* __restrict__ blog, bf16* __restrict__ cT) {
    int tid = blockIdx.x * blockDim.x + threadIdx.x;
    if (tid >= CB * CI) return;
    int b = tid / CI, i = tid - b * CI;
    const float* row = blog + (long)tid * CO;
    float mx = -3.4e38f;
    for (int oo = 0; oo < CO; ++oo) mx = fmaxf(mx, row[oo]);
    float sum = 0.f;
    for (int oo = 0; oo < CO; ++oo) sum += __expf(row[oo] - mx);
    float inv = 1.0f / sum;
    for (int oo = 0; oo < CO; ++oo)
        cT[((long)oo * CI + i) * CB + b] = (bf16)(__expf(row[oo] - mx) * inv);
}

// reduce partials, squash, emit v (fp32 to d_out, bf16 [O][B][VO] for next pass)
__global__ void caps_squash(const float* __restrict__ s_part,
                            float* __restrict__ out, bf16* __restrict__ vbf) {
    int tid = blockIdx.x * blockDim.x + threadIdx.x;
    if (tid >= CB * CO) return;
    int b = tid / CO, o = tid - b * CO;
    float s[CVO];
    #pragma unroll
    for (int v = 0; v < CVO; ++v) s[v] = 0.f;
    for (int p = 0; p < NPART; ++p) {
        const float* sp = s_part + ((long)(o * NPART + p) * CB + b) * CVO;
        #pragma unroll
        for (int v = 0; v < CVO; ++v) s[v] += sp[v];
    }
    float sq = 0.f;
    #pragma unroll
    for (int v = 0; v < CVO; ++v) sq += s[v] * s[v];
    float scale = (sq / (1.0f + sq)) * rsqrtf(sq + 1e-8f);
    #pragma unroll
    for (int v = 0; v < CVO; ++v) {
        float val = s[v] * scale;
        out[((long)b * CO + o) * CVO + v] = val;
        vbf[((long)o * CB + b) * CVO + v] = (bf16)val;
    }
}

extern "C" void kernel_launch(void* const* d_in, const int* in_sizes, int n_in,
                              void* d_out, int out_size, void* d_ws, size_t ws_size,
                              hipStream_t stream) {
    const float* x = (const float*)d_in[0];   // [32,2048,16]
    const float* W = (const float*)d_in[1];   // [64,2048,32,16]
    float* out = (float*)d_out;               // [32,64,32]

    char* ws = (char*)d_ws;
    size_t off = 0;
    auto carve = [&](size_t bytes) { char* p = ws + off; off = (off + bytes + 255) & ~(size_t)255; return p; };
    bf16*  Wbf    = (bf16*)carve((size_t)CO * CI * CVO * CVI * 2);   // 134 MB
    bf16*  WT     = (bf16*)carve((size_t)CO * CI * CVO * CVI * 2);   // 134 MB
    bf16*  x1     = (bf16*)carve((size_t)CB * CI * CVI * 2);         // 2 MB
    float* blog   = (float*)carve((size_t)CB * CI * CO * 4);         // 16 MB
    bf16*  cT     = (bf16*)carve((size_t)CO * CI * CB * 2);          // 8 MB
    float* s_part = (float*)carve((size_t)CO * NPART * CB * CVO * 4);// 4 MB
    bf16*  vbf    = (bf16*)carve((size_t)CO * CB * CVO * 2);         // 128 KB

    caps_conv_w<<<8192, 256, 0, stream>>>(W, Wbf, WT);
    caps_conv_x<<<512, 256, 0, stream>>>(x, x1);

    // iter 0: c uniform = 1/64 (scalar pulled out of the GEMM)
    caps_spass<1><<<CO * NPART, 32, 0, stream>>>(Wbf, x1, cT, s_part);
    caps_squash<<<(CB * CO + 255) / 256, 256, 0, stream>>>(s_part, out, vbf);

    // iter 1: agreement (overwrite logits) -> softmax -> s -> squash
    caps_apass<0><<<CO * NPART, 32, 0, stream>>>(WT, x1, vbf, blog);
    caps_softmax<<<(CB * CI + 255) / 256, 256, 0, stream>>>(blog, cT);
    caps_spass<0><<<CO * NPART, 32, 0, stream>>>(Wbf, x1, cT, s_part);
    caps_squash<<<(CB * CO + 255) / 256, 256, 0, stream>>>(s_part, out, vbf);

    // iter 2: agreement (accumulate logits) -> softmax -> s -> squash
    caps_apass<1><<<CO * NPART, 32, 0, stream>>>(WT, x1, vbf, blog);
    caps_softmax<<<(CB * CI + 255) / 256, 256, 0, stream>>>(blog, cT);
    caps_spass<0><<<CO * NPART, 32, 0, stream>>>(Wbf, x1, cT, s_part);
    caps_squash<<<(CB * CO + 255) / 256, 256, 0, stream>>>(s_part, out, vbf);
}
